// LocalSelfAttention_53644141527453
// MI455X (gfx1250) — matrix operand order repaired and checked by reading, not compiled, not run on hardware
//
#include <hip/hip_runtime.h>
#include <hip/hip_bf16.h>
#include <stdint.h>

// ---------------------------------------------------------------------------
// Local (sliding-window) self-attention for MI455X / gfx1250, wave32 + WMMA.
// cast->bf16, double-buffered WMMA GEMM (QKV, V stored transposed),
// flash-style windowed attention (online softmax), WMMA GEMM (out proj).
// ---------------------------------------------------------------------------

typedef __attribute__((ext_vector_type(16))) __bf16        v16bf;
typedef __attribute__((ext_vector_type(8)))  float         v8f;
typedef __attribute__((ext_vector_type(4)))  unsigned int  u32x4;
typedef unsigned short u16;

#define T_SEQ   2048
#define C_DIM   1024
#define N_HEAD  16
#define D_K     64
#define WIN     128
#define BATCH   4
#define M_ROWS  (BATCH * T_SEQ)     /* 8192 */
#define N_QKV   (3 * C_DIM)         /* 3072 */
#define TP      2336                /* padded Vt row: key k stored at k+WIN  */

#define BM 128
#define BN 256
#define LDP 40                      /* padded LDS row stride (u16) for 32 cols */
#define A_BUF (BM * LDP)            /* u16 elems per A buffer */
#define B_BUF (BN * LDP)

struct B32x8 { u32x4 lo, hi; };

static __device__ __forceinline__ v16bf mkfrag(u32x4 lo, u32x4 hi) {
  B32x8 t; t.lo = lo; t.hi = hi;
  return __builtin_bit_cast(v16bf, t);
}

static __device__ __forceinline__ u16 f2bf(float f) {
  unsigned u = __builtin_bit_cast(unsigned, f);
  u += 0x7FFFu + ((u >> 16) & 1u);           // round-to-nearest-even
  return (u16)(u >> 16);
}

static __device__ __forceinline__ float rmax16(float v) {
  v = fmaxf(v, __shfl_xor(v, 1, 32));
  v = fmaxf(v, __shfl_xor(v, 2, 32));
  v = fmaxf(v, __shfl_xor(v, 4, 32));
  v = fmaxf(v, __shfl_xor(v, 8, 32));
  return v;                                   // masks < 16 stay in 16-lane half
}
static __device__ __forceinline__ float rsum16(float v) {
  v += __shfl_xor(v, 1, 32);
  v += __shfl_xor(v, 2, 32);
  v += __shfl_xor(v, 4, 32);
  v += __shfl_xor(v, 8, 32);
  return v;
}

// ---------------------------------------------------------------------------
// helpers: cast / zero
// ---------------------------------------------------------------------------
__global__ void k_cast_bf16(const float* __restrict__ in, u16* __restrict__ out, int n) {
  int i = blockIdx.x * blockDim.x + threadIdx.x;
  if (i < n) out[i] = f2bf(in[i]);
}

__global__ void k_zero_u16(u16* __restrict__ p, int n) {
  int i = blockIdx.x * blockDim.x + threadIdx.x;
  if (i < n) p[i] = 0;
}

// ---------------------------------------------------------------------------
// GEMM staging: issue ALL global loads into distinct regs (overlapped), store
// to LDS later. A tile BMx32 (2 chunks/thread), B tile BNx32 (4 chunks/thread).
// ---------------------------------------------------------------------------
static __device__ __forceinline__ void load_stage(
    const u16* __restrict__ A, const u16* __restrict__ Bw, int K,
    int bm, int bn, int k0, u32x4 ra[2], u32x4 rb[4])
{
  const int tid = threadIdx.x;
#pragma unroll
  for (int i = 0; i < 2; ++i) {
    int c = tid + i * 256, r = c >> 2, cc = (c & 3) * 8;
    ra[i] = *(const u32x4*)(A + (size_t)(bm + r) * K + k0 + cc);
  }
#pragma unroll
  for (int i = 0; i < 4; ++i) {
    int c = tid + i * 256, r = c >> 2, cc = (c & 3) * 8;
    rb[i] = *(const u32x4*)(Bw + (size_t)(bn + r) * K + k0 + cc);
  }
}

static __device__ __forceinline__ void store_stage(
    u16* lA, u16* lB, const u32x4 ra[2], const u32x4 rb[4])
{
  const int tid = threadIdx.x;
#pragma unroll
  for (int i = 0; i < 2; ++i) {
    int c = tid + i * 256, r = c >> 2, cc = (c & 3) * 8;
    *(u32x4*)(lA + r * LDP + cc) = ra[i];
  }
#pragma unroll
  for (int i = 0; i < 4; ++i) {
    int c = tid + i * 256, r = c >> 2, cc = (c & 3) * 8;
    *(u32x4*)(lB + r * LDP + cc) = rb[i];
  }
}

// 16 WMMAs per wave per K-step: wave tile 64x64, B-frags streamed.
static __device__ __forceinline__ void compute_step(
    const u16* lA, const u16* lB, int wm, int wn, int sel, int l15,
    v8f acc[4][4])
{
  v16bf af[4];
#pragma unroll
  for (int i = 0; i < 4; ++i) {   // A frag: lane=row, 16 contiguous K
    const u16* p = lA + (wm + i * 16 + l15) * LDP;
    af[i] = mkfrag(*(const u32x4*)(p + sel * 16),
                   *(const u32x4*)(p + sel * 16 + 8));
  }
#pragma unroll
  for (int j = 0; j < 4; ++j) {   // B frag: lane=col(N), 16 contiguous K
    const u16* p = lB + (wn + j * 16 + l15) * LDP + sel * 16;
    v16bf bf = mkfrag(*(const u32x4*)p, *(const u32x4*)(p + 8));
#pragma unroll
    for (int i = 0; i < 4; ++i)
      acc[i][j] = __builtin_amdgcn_wmma_f32_16x16x32_bf16(
          false, af[i], false, bf, (short)0, acc[i][j], false, false);
  }
}

// Double-buffered mainloop: one barrier per K-step, loads overlap 16 WMMAs.
static __device__ __forceinline__ void gemm_tiles(
    const u16* __restrict__ A, const u16* __restrict__ Bw, int K,
    int bm, int bn, u16* lA, u16* lB, v8f acc[4][4])
{
  const int lane = threadIdx.x & 31, wid = threadIdx.x >> 5;
  const int wm = (wid >> 2) * 64, wn = (wid & 3) * 64;
  const int sel = lane >> 4, l15 = lane & 15;
  const int NS = K / 32;

  u32x4 ra[2], rb[4];
  load_stage(A, Bw, K, bm, bn, 0, ra, rb);
  store_stage(lA, lB, ra, rb);

  for (int s = 0; s < NS; ++s) {
    __syncthreads();
    const int cur = s & 1, nxt = cur ^ 1;
    if (s + 1 < NS) {
      load_stage(A, Bw, K, bm, bn, (s + 1) * 32, ra, rb);   // in flight
      if (s + 2 < NS) {                                     // L2 prefetch
        const int tid = threadIdx.x;
        __builtin_prefetch(A  + (size_t)(bm + (tid >> 1)) * K + (s + 2) * 32 + (tid & 1) * 16, 0, 1);
        __builtin_prefetch(Bw + (size_t)(bn + (tid >> 2)) * K + (s + 2) * 32 + (tid & 3) * 8, 0, 1);
      }
    }
    compute_step(lA + cur * A_BUF, lB + cur * B_BUF, wm, wn, sel, l15, acc);
    if (s + 1 < NS)
      store_stage(lA + nxt * A_BUF, lB + nxt * B_BUF, ra, rb);
  }
}

// ---------------------------------------------------------------------------
// GEMM 1: qkv = x @ w_qkv^T ; scatter into Q,K [B,H,T,DK] and Vt [B,H,DK,TP]
// ---------------------------------------------------------------------------
__global__ __launch_bounds__(256) void k_gemm_qkv(
    const u16* __restrict__ A, const u16* __restrict__ Bw,
    u16* __restrict__ Q, u16* __restrict__ Kk, u16* __restrict__ Vt)
{
  __shared__ u16 lA[2 * A_BUF];
  __shared__ u16 lB[2 * B_BUF];
  const int bm = blockIdx.y * BM, bn = blockIdx.x * BN;
  const int lane = threadIdx.x & 31, wid = threadIdx.x >> 5;
  const int sel = lane >> 4, l15 = lane & 15;
  const int wm = (wid >> 2) * 64, wn = (wid & 3) * 64;

  v8f acc[4][4];
  v8f zero = {0.f, 0.f, 0.f, 0.f, 0.f, 0.f, 0.f, 0.f};
#pragma unroll
  for (int i = 0; i < 4; ++i)
#pragma unroll
    for (int j = 0; j < 4; ++j) acc[i][j] = zero;

  gemm_tiles(A, Bw, C_DIM, bm, bn, lA, lB, acc);

#pragma unroll
  for (int i = 0; i < 4; ++i)
#pragma unroll
    for (int j = 0; j < 4; ++j) {
      const int nb = bn + wn + j * 16;           // tile-uniform
      const int part = nb >> 10;                 // 0:q 1:k 2:v
      const int c = (nb + l15) & 1023;
      const int h = c >> 6, d = c & 63;
#pragma unroll
      for (int v = 0; v < 8; ++v) {
        const int m = bm + wm + i * 16 + v + 8 * sel;
        const int b = m >> 11, t = m & 2047;
        const size_t bh = (size_t)(b * N_HEAD + h);
        const u16 hv = f2bf(acc[i][j][v]);
        if (part == 0)      Q [(bh * T_SEQ + t) * D_K + d] = hv;
        else if (part == 1) Kk[(bh * T_SEQ + t) * D_K + d] = hv;
        else                Vt[(bh * D_K + d) * TP + t + WIN] = hv;
      }
    }
}

// ---------------------------------------------------------------------------
// GEMM 2: out = attn @ w_proj^T -> f32
// ---------------------------------------------------------------------------
__global__ __launch_bounds__(256) void k_gemm_proj(
    const u16* __restrict__ A, const u16* __restrict__ Bw, float* __restrict__ out)
{
  __shared__ u16 lA[2 * A_BUF];
  __shared__ u16 lB[2 * B_BUF];
  const int bm = blockIdx.y * BM, bn = blockIdx.x * BN;
  const int lane = threadIdx.x & 31, wid = threadIdx.x >> 5;
  const int sel = lane >> 4, l15 = lane & 15;
  const int wm = (wid >> 2) * 64, wn = (wid & 3) * 64;

  v8f acc[4][4];
  v8f zero = {0.f, 0.f, 0.f, 0.f, 0.f, 0.f, 0.f, 0.f};
#pragma unroll
  for (int i = 0; i < 4; ++i)
#pragma unroll
    for (int j = 0; j < 4; ++j) acc[i][j] = zero;

  gemm_tiles(A, Bw, C_DIM, bm, bn, lA, lB, acc);

#pragma unroll
  for (int i = 0; i < 4; ++i)
#pragma unroll
    for (int j = 0; j < 4; ++j) {
      const int n = bn + wn + j * 16 + l15;
#pragma unroll
      for (int v = 0; v < 8; ++v) {
        const int m = bm + wm + i * 16 + v + 8 * sel;
        out[(size_t)m * C_DIM + n] = acc[i][j][v];
      }
    }
}

// ---------------------------------------------------------------------------
// Windowed attention: one wave handles (b,h, 16-query tile); online softmax.
// P relayout uses a private per-wave LDS region -> only an intra-wave DS
// ordering fence is needed (LDS is in-order per wave), no workgroup barrier.
// ---------------------------------------------------------------------------
__global__ __launch_bounds__(128) void k_attn(
    const u16* __restrict__ Q, const u16* __restrict__ Kk,
    const u16* __restrict__ Vt, u16* __restrict__ Att)
{
  __shared__ u16 ldsP[4][16 * LDP];             // per-wave P bounce buffer

  const int lane = threadIdx.x & 31, wid = threadIdx.x >> 5;
  const int sel = lane >> 4, l15 = lane & 15;
  const int idx = blockIdx.x * 4 + wid;
  const int qt = idx & 127, h = (idx >> 7) & 15, b = idx >> 11;
  const int t0 = qt * 16;

  const size_t bh    = (size_t)(b * N_HEAD + h);
  const size_t qbase = (bh * T_SEQ + t0) * D_K;
  const size_t kbase = bh * T_SEQ * D_K;
  const size_t vbase = bh * D_K * TP;

  // Q fragments (DK=64 -> two K=32 A-frags)
  v16bf qf[2];
#pragma unroll
  for (int kk = 0; kk < 2; ++kk) {
    const u16* p = Q + qbase + (size_t)l15 * D_K + kk * 32;
    qf[kk] = mkfrag(*(const u32x4*)(p + sel * 16),
                    *(const u32x4*)(p + sel * 16 + 8));
  }

  float mrun[8], lrun[8];
  v8f o[4];
  v8f zero = {0.f, 0.f, 0.f, 0.f, 0.f, 0.f, 0.f, 0.f};
#pragma unroll
  for (int v = 0; v < 8; ++v) { mrun[v] = -1e30f; lrun[v] = 0.f; }
#pragma unroll
  for (int d = 0; d < 4; ++d) o[d] = zero;

  for (int kt = 0; kt < 9; ++kt) {              // 9 * 32 keys cover [-128,+159]
    const int k0 = t0 - WIN + kt * 32;

    // ---- S = Q*K^T over this 32-key tile (two 16-col S tiles) ----
    v8f s[2];
#pragma unroll
    for (int j = 0; j < 2; ++j) {
      const int key = k0 + j * 16 + l15;
      const int kc = key < 0 ? 0 : (key > T_SEQ - 1 ? T_SEQ - 1 : key);
      const u16* kp = Kk + kbase + (size_t)kc * D_K;
      s[j] = zero;
#pragma unroll
      for (int kk = 0; kk < 2; ++kk) {
        v16bf kf = mkfrag(*(const u32x4*)(kp + kk * 32 + sel * 16),
                          *(const u32x4*)(kp + kk * 32 + sel * 16 + 8));
        s[j] = __builtin_amdgcn_wmma_f32_16x16x32_bf16(
            false, qf[kk], false, kf, (short)0, s[j], false, false);
      }
    }

    // ---- mask + scale + online softmax update ----
    float fsc[8];
#pragma unroll
    for (int v = 0; v < 8; ++v) {
      const int qq = t0 + v + 8 * sel;
      float sv[2];
#pragma unroll
      for (int j = 0; j < 2; ++j) {
        const int key = k0 + j * 16 + l15;
        const int dlt = qq - key;
        const bool valid = (key >= 0) && (key < T_SEQ) && (dlt <= WIN) && (dlt >= -WIN);
        sv[j] = valid ? s[j][v] * 0.125f : -1e30f;
      }
      const float rm = rmax16(fmaxf(sv[0], sv[1]));
      const float mnew = fmaxf(fmaxf(mrun[v], rm), -1e20f);
      fsc[v] = __expf(mrun[v] - mnew);
      const float p0 = __expf(sv[0] - mnew);
      const float p1 = __expf(sv[1] - mnew);
      lrun[v] = lrun[v] * fsc[v] + rsum16(p0 + p1);
      mrun[v] = mnew;
      // C-layout -> LDS (row-major 16x32, pad 8) for A-frag reload
      const int row = v + 8 * sel;
      ldsP[wid][row * LDP + l15]      = f2bf(p0);
      ldsP[wid][row * LDP + 16 + l15] = f2bf(p1);
    }
#pragma unroll
    for (int d = 0; d < 4; ++d)
#pragma unroll
      for (int v = 0; v < 8; ++v) o[d][v] *= fsc[v];

    // intra-wave LDS RAW fence (per-wave region; DS is in-order per wave)
    asm volatile("s_wait_dscnt 0" ::: "memory");

    // P as A-fragment (16 queries x 32 keys)
    const u16* pp = &ldsP[wid][l15 * LDP];
    v16bf pf = mkfrag(*(const u32x4*)(pp + sel * 16),
                      *(const u32x4*)(pp + sel * 16 + 8));

    // ---- O += P * V ; Vt rows are contiguous in key -> contiguous B-frags
#pragma unroll
    for (int dg = 0; dg < 4; ++dg) {
      const u16* vp = Vt + vbase + (size_t)(dg * 16 + l15) * TP
                      + (k0 + WIN + sel * 16);
      v16bf vf = mkfrag(*(const u32x4*)vp, *(const u32x4*)(vp + 8));
      o[dg] = __builtin_amdgcn_wmma_f32_16x16x32_bf16(
          false, pf, false, vf, (short)0, o[dg], false, false);
    }
  }

  // ---- normalize + store attention output (bf16, [B*T, C]) ----
  float inv[8];
#pragma unroll
  for (int v = 0; v < 8; ++v) inv[v] = 1.f / lrun[v];
#pragma unroll
  for (int dg = 0; dg < 4; ++dg)
#pragma unroll
    for (int v = 0; v < 8; ++v) {
      const int t = t0 + v + 8 * sel;
      const int c = h * D_K + dg * 16 + l15;
      Att[(size_t)(b * T_SEQ + t) * C_DIM + c] = f2bf(o[dg][v] * inv[v]);
    }
}

// ---------------------------------------------------------------------------
// launch
// ---------------------------------------------------------------------------
extern "C" void kernel_launch(void* const* d_in, const int* in_sizes, int n_in,
                              void* d_out, int out_size, void* d_ws, size_t ws_size,
                              hipStream_t stream) {
  (void)in_sizes; (void)n_in; (void)out_size; (void)ws_size;

  const float* x      = (const float*)d_in[0];   // [4,2048,1024]
  const float* w_qkv  = (const float*)d_in[1];   // [3072,1024]
  const float* w_proj = (const float*)d_in[2];   // [1024,1024]
  float* out = (float*)d_out;                    // [4,2048,1024]

  char* ws = (char*)d_ws;
  const size_t nX   = (size_t)M_ROWS * C_DIM;        // 8388608
  const size_t nWq  = (size_t)N_QKV * C_DIM;         // 3145728
  const size_t nWp  = (size_t)C_DIM * C_DIM;         // 1048576
  const size_t nQKV = (size_t)M_ROWS * C_DIM;        // per Q/K buffer
  const size_t nVt  = (size_t)BATCH * N_HEAD * D_K * TP;

  u16* xb  = (u16*)(ws);
  u16* wqb = (u16*)(ws + nX * 2);
  u16* wpb = (u16*)(ws + (nX + nWq) * 2);
  u16* Qb  = (u16*)(ws + (nX + nWq + nWp) * 2);
  u16* Kb  = (u16*)(ws + (nX + nWq + nWp + nQKV) * 2);
  u16* Vtb = (u16*)(ws + (nX + nWq + nWp + 2 * nQKV) * 2);
  u16* Ab  = (u16*)(ws + (nX + nWq + nWp + 2 * nQKV + nVt) * 2);

  // 1) casts + Vt pad zeroing
  k_cast_bf16<<<dim3((nX  + 255) / 256), 256, 0, stream>>>(x, xb, (int)nX);
  k_cast_bf16<<<dim3((nWq + 255) / 256), 256, 0, stream>>>(w_qkv, wqb, (int)nWq);
  k_cast_bf16<<<dim3((nWp + 255) / 256), 256, 0, stream>>>(w_proj, wpb, (int)nWp);
  k_zero_u16 <<<dim3((nVt + 255) / 256), 256, 0, stream>>>(Vtb, (int)nVt);

  // 2) QKV projection (M=8192, N=3072, K=1024)
  k_gemm_qkv<<<dim3(N_QKV / BN, M_ROWS / BM), 256, 0, stream>>>(xb, wqb, Qb, Kb, Vtb);

  // 3) windowed attention: 8192 (b,h,qtile) waves, 4 waves / block
  k_attn<<<dim3(M_ROWS * N_HEAD / 16 / 4), 128, 0, stream>>>(Qb, Kb, Vtb, Ab);

  // 4) output projection (M=8192, N=1024, K=1024)
  k_gemm_proj<<<dim3(C_DIM / BN, M_ROWS / BM), 256, 0, stream>>>(Ab, wpb, out);
}
